// C3STR_20581483282725
// MI455X (gfx1250) — matrix operand order repaired and checked by reading, not compiled, not run on hardware
//
#include <hip/hip_runtime.h>
#include <math.h>
#include <stdint.h>

// ---------------------------------------------------------------------------
// CDNA5 (gfx1250, wave32) WMMA types + async-copy helpers
// ---------------------------------------------------------------------------
typedef __attribute__((ext_vector_type(16))) __bf16 v16bf;
typedef __attribute__((ext_vector_type(8)))  __bf16 v8bf;
typedef __attribute__((ext_vector_type(8)))  float  v8f;

__device__ __forceinline__ __bf16 bfc(float x) { return (__bf16)x; }

// Low 32 bits of a generic pointer to LDS = LDS byte address (ISA 10.2).
__device__ __forceinline__ unsigned lds_off(const void* p) {
  return (unsigned)(uintptr_t)p;
}
// GLOBAL_LOAD_ASYNC_TO_LDS_B128: per-lane 16B global -> 16B LDS, ASYNCcnt.
__device__ __forceinline__ void async_g2l_b128(unsigned lds, const void* g) {
  asm volatile("global_load_async_to_lds_b128 %0, %1, off"
               :: "v"(lds), "v"((unsigned long long)(uintptr_t)g)
               : "memory");
}
// Partial waits: ASYNCcnt decrements in issue order, so allowing N outstanding
// retires everything older than the newest N async instructions.
__device__ __forceinline__ void wait_async0()   { asm volatile("s_wait_asynccnt 0x0" ::: "memory"); }
__device__ __forceinline__ void wait_async_le2(){ asm volatile("s_wait_asynccnt 0x2" ::: "memory"); }
__device__ __forceinline__ void wait_async_le4(){ asm volatile("s_wait_asynccnt 0x4" ::: "memory"); }

// A-fragment (16x32 bf16, MxK): lane<16 -> row=lane, K base 0; lane>=16 ->
// row=lane-16, base 8. Element j: K = base + (j<8 ? j : j+8)   (ISA 7.12.2)
__device__ __forceinline__ v16bf load_a_frag(const __bf16* row, int lane) {
  const int kb = (lane < 16) ? 0 : 8;
  v8bf lo = *(const v8bf*)(row + kb);
  v8bf hi = *(const v8bf*)(row + kb + 16);
  v16bf a;
#pragma unroll
  for (int j = 0; j < 8; ++j) { a[j] = lo[j]; a[j + 8] = hi[j]; }
  return a;
}
// B-fragment (32x16 bf16, KxN): lane<16 -> col=lane, K base 0; lane>=16 ->
// col=lane-16, base 8. Element j: K = base + (j<8 ? j : j+8) (matches A).
__device__ __forceinline__ v16bf load_b_frag(const __bf16* col, int lane) {
  const int kb = (lane < 16) ? 0 : 8;
  v8bf lo = *(const v8bf*)(col + kb);
  v8bf hi = *(const v8bf*)(col + kb + 16);
  v16bf b;
#pragma unroll
  for (int j = 0; j < 8; ++j) { b[j] = lo[j]; b[j + 8] = hi[j]; }
  return b;
}

// ---------------------------------------------------------------------------
// fp32 -> bf16 conversion (weights / input image), grid-stride
// ---------------------------------------------------------------------------
__global__ __launch_bounds__(256) void f2b_kernel(const float* __restrict__ in,
                                                  __bf16* __restrict__ outp,
                                                  long n) {
  long i = (long)blockIdx.x * 256 + threadIdx.x;
  const long stride = (long)gridDim.x * 256;
  for (; i < n; i += stride) outp[i] = (__bf16)in[i];
}

// ---------------------------------------------------------------------------
// Strided bf16 GEMM:  C[b,m,n] = epilogue( sum_k A[m,k] * B(b,k,n) )
// A: MxK row-major bf16.  B: bf16 via strides, optional 2nd source for
// k >= Ksplit (cv3 channel concat).  bpath: 0 = token-major (contig k, async
// staging), 1 = planar (contig n, vector+transpose), 2 = generic/split.
// Epilogues: 0 BN+SiLU, 1 +bias, 2 +bias,GELU, 3 +bias+residual.
// Outputs: optional fp32 (Cf) and/or bf16 (Cb), same strides.
// Double-buffered LDS: next tile's async DMA overlaps current tile's WMMAs.
// ---------------------------------------------------------------------------
struct GemmP {
  const __bf16* A;
  const __bf16* B0;
  const __bf16* B1;
  float*        Cf;
  __bf16*       Cb;
  const float*  res;
  const float*  e0;
  long b0b, b0k, b0n;
  long b1b, b1k, b1n;
  long cb, cm, cn;
  int  M, N, K, Ksplit, Mtot, mode, bpath;
};

#define BM 128
#define BN 128
#define BKK 32

__global__ __launch_bounds__(256) void gemm_wmma_bf16(GemmP p) {
  __shared__ alignas(16) __bf16 As[2][BM][BKK + 8];   // [m][k], row stride 80B
  __shared__ alignas(16) __bf16 Bs[2][BN][BKK + 8];   // [n][k]
  const int tid  = threadIdx.x;
  const int lane = tid & 31;
  const int wave = tid >> 5;
  const int bm0 = blockIdx.y * BM, bn0 = blockIdx.x * BN, batch = blockIdx.z;
  const int wm = (wave >> 2) * 64;   // wave sub-tile 64x32 -> 8 accumulators
  const int wn = (wave & 3) * 32;

  v8f acc[4][2];
#pragma unroll
  for (int i = 0; i < 4; ++i)
#pragma unroll
    for (int j = 0; j < 2; ++j)
#pragma unroll
      for (int r = 0; r < 8; ++r) acc[i][j][r] = 0.0f;

  // stage tile at K offset k0 into LDS buffer `buf`
  auto stage = [&](int k0, int buf) {
    {  // A: 2 x 16B async chunks per thread
      const int c = (tid & 3) * 8;
      int r = (tid >> 2);
      async_g2l_b128(lds_off(&As[buf][r][c]),
                     p.A + (size_t)(bm0 + r) * p.K + k0 + c);
      r += 64;
      async_g2l_b128(lds_off(&As[buf][r][c]),
                     p.A + (size_t)(bm0 + r) * p.K + k0 + c);
    }
    if (p.bpath == 0) {              // token-major: contiguous k, async DMA
      const int c = (tid & 3) * 8;
      int n = (tid >> 2);
      async_g2l_b128(lds_off(&Bs[buf][n][c]),
                     p.B0 + (size_t)batch * p.b0b + (size_t)(bn0 + n) * p.b0n + k0 + c);
      n += 64;
      async_g2l_b128(lds_off(&Bs[buf][n][c]),
                     p.B0 + (size_t)batch * p.b0b + (size_t)(bn0 + n) * p.b0n + k0 + c);
    } else if (p.bpath == 1) {       // planar: contiguous n, vector + transpose
#pragma unroll
      for (int it = 0; it < 2; ++it) {
        const int k = (tid >> 4) + it * 16;
        const int n = (tid & 15) * 8;
        const v8bf c8 = *(const v8bf*)(p.B0 + (size_t)batch * p.b0b +
                                       (size_t)(k0 + k) * p.b0k + (bn0 + n));
#pragma unroll
        for (int u = 0; u < 8; ++u) Bs[buf][n + u][k] = c8[u];
      }
    } else {                         // generic / concat split (cv3)
#pragma unroll
      for (int it = 0; it < 4; ++it) {
        const int n  = (tid >> 3) + it * 32;
        const int kc = (tid & 7) * 4;
#pragma unroll
        for (int u = 0; u < 4; ++u) {
          const int k = k0 + kc + u;
          __bf16 v;
          if (p.B1 && k >= p.Ksplit)
            v = p.B1[(size_t)batch * p.b1b + (size_t)(k - p.Ksplit) * p.b1k +
                     (size_t)(bn0 + n) * p.b1n];
          else
            v = p.B0[(size_t)batch * p.b0b + (size_t)k * p.b0k +
                     (size_t)(bn0 + n) * p.b0n];
          Bs[buf][n][kc + u] = v;
        }
      }
    }
  };

  stage(0, 0);
  const int nsteps = p.K / BKK;
  for (int s = 0; s < nsteps; ++s) {
    const int cur = s & 1;
    const bool havenext = (s + 1 < nsteps);
    if (havenext) stage((s + 1) * BKK, cur ^ 1);   // overlaps with WMMAs below
    // retire only this buffer's async copies; next tile's stay in flight
    if (!havenext)      wait_async0();
    else if (p.bpath == 0) wait_async_le4();
    else                wait_async_le2();
    __syncthreads();    // all waves' staging for `cur` visible

    v16bf afr[4];
#pragma unroll
    for (int mt = 0; mt < 4; ++mt)
      afr[mt] = load_a_frag(&As[cur][wm + mt * 16 + (lane & 15)][0], lane);
    v16bf bfr[2];
#pragma unroll
    for (int nt = 0; nt < 2; ++nt)
      bfr[nt] = load_b_frag(&Bs[cur][wn + nt * 16 + (lane & 15)][0], lane);
#pragma unroll
    for (int mt = 0; mt < 4; ++mt)
#pragma unroll
      for (int nt = 0; nt < 2; ++nt)
        acc[mt][nt] = __builtin_amdgcn_wmma_f32_16x16x32_bf16(
            false, afr[mt], false, bfr[nt], (short)0, acc[mt][nt], false, false);
    __syncthreads();    // fragment reads done before `cur` is re-staged
  }

  // ---- fused epilogue (C-frag: m = r|r+8, n = lane&15) ----
#pragma unroll
  for (int mt = 0; mt < 4; ++mt)
#pragma unroll
    for (int nt = 0; nt < 2; ++nt)
#pragma unroll
      for (int r = 0; r < 8; ++r) {
        const int m = bm0 + wm + mt * 16 + ((lane < 16) ? r : r + 8);
        const int n = bn0 + wn + nt * 16 + (lane & 15);
        const size_t off = (size_t)batch * p.cb + (size_t)m * p.cm +
                           (size_t)n * p.cn;
        float v = acc[mt][nt][r];
        if (p.mode == 0) {
          const float g  = p.e0[m];
          const float bb = p.e0[p.Mtot + m];
          const float mu = p.e0[2 * p.Mtot + m];
          const float vr = p.e0[3 * p.Mtot + m];
          const float s  = g * rsqrtf(vr + 1e-5f);
          v = v * s + (bb - mu * s);
          v = v / (1.0f + __expf(-v));                               // SiLU
        } else if (p.mode == 1) {
          v += p.e0[m];
        } else if (p.mode == 2) {
          v += p.e0[m];
          v = 0.5f * v * (1.0f + erff(v * 0.70710678118654752f));    // GELU
        } else {
          v += p.e0[m] + p.res[off];
        }
        if (p.Cf) p.Cf[off] = v;
        if (p.Cb) p.Cb[off] = (__bf16)v;
      }
}

// ---------------------------------------------------------------------------
// LayerNorm over C=256 (fp32 in, bf16 out), one block per token
// ---------------------------------------------------------------------------
__global__ __launch_bounds__(256) void ln_kernel(const float* __restrict__ X,
                                                 __bf16* __restrict__ Y,
                                                 const float* __restrict__ g,
                                                 const float* __restrict__ b) {
  __shared__ float red[256];
  const size_t t = blockIdx.x;
  const int c = threadIdx.x;
  const float v = X[t * 256 + c];
  red[c] = v;
  __syncthreads();
  for (int s = 128; s > 0; s >>= 1) {
    if (c < s) red[c] += red[c + s];
    __syncthreads();
  }
  const float mu = red[0] * (1.0f / 256.0f);
  __syncthreads();
  const float d = v - mu;
  red[c] = d * d;
  __syncthreads();
  for (int s = 128; s > 0; s >>= 1) {
    if (c < s) red[c] += red[c + s];
    __syncthreads();
  }
  const float var = red[0] * (1.0f / 256.0f);
  Y[t * 256 + c] = (__bf16)(d * rsqrtf(var + 1e-5f) * g[c] + b[c]);
}

// ---------------------------------------------------------------------------
// Swin window attention: block per (window, head), 4 waves, all-WMMA.
// qkv is bf16 token-major (768/token); q/k staged via per-lane async DMA
// (gather addresses), v transposed manually. Softmax folds the 1/sqrt(hd)
// scale, rel-pos bias, and analytic SW-MSA region mask.
// ---------------------------------------------------------------------------
__global__ __launch_bounds__(128) void swin_attn(const __bf16* __restrict__ qkv,
                                                 __bf16* __restrict__ outp,
                                                 const float* __restrict__ rpb,
                                                 int shift) {
  __shared__ alignas(16) __bf16 qs[64][40];    // [p][d]
  __shared__ alignas(16) __bf16 ks[64][40];    // [m][d]
  __shared__ alignas(16) __bf16 vts[32][72];   // [d][q] (v transposed)
  __shared__ float att[64][65];
  __shared__ alignas(16) __bf16 attb[64][72];
  __shared__ int tok_s[64];
  __shared__ int cnt_s[64];

  const int tid = threadIdx.x, lane = tid & 31, wave = tid >> 5;
  const int win = blockIdx.x, head = blockIdx.y;
  const int b = win >> 6, widx = win & 63, wy = widx >> 3, wx = widx & 7;

  if (tid < 64) {
    const int i = tid >> 3, j = tid & 7;
    const int hr = wy * 8 + i, wr = wx * 8 + j;          // rolled coords
    tok_s[tid] = b * 4096 + ((hr + shift) & 63) * 64 + ((wr + shift) & 63);
    const int rh = (hr < 56) ? 0 : ((hr < 64 - shift) ? 1 : 2);
    const int rw = (wr < 56) ? 0 : ((wr < 64 - shift) ? 1 : 2);
    cnt_s[tid] = rh * 3 + rw;
  }

  // q/k rows: 64 rows x 4 chunks of 16B each -> 2 async DMAs per thread each
#pragma unroll
  for (int it = 0; it < 2; ++it) {
    const int idx = tid + it * 128;      // 0..255
    const int pp = idx >> 2, c = (idx & 3) * 8;
    const int hr = wy * 8 + (pp >> 3), wr = wx * 8 + (pp & 7);
    const size_t tok = (size_t)b * 4096 + ((hr + shift) & 63) * 64 +
                       ((wr + shift) & 63);
    const size_t base = tok * 768 + head * 32 + c;
    async_g2l_b128(lds_off(&qs[pp][c]), qkv + base);
    async_g2l_b128(lds_off(&ks[pp][c]), qkv + base + 256);
  }
  // v transposed: 2048 scalars
  for (int idx = tid; idx < 64 * 32; idx += 128) {
    const int pp = idx >> 5, d = idx & 31;
    const int hr = wy * 8 + (pp >> 3), wr = wx * 8 + (pp & 7);
    const size_t tok = (size_t)b * 4096 + ((hr + shift) & 63) * 64 +
                       ((wr + shift) & 63);
    vts[d][pp] = qkv[tok * 768 + head * 32 + d + 512];
  }
  wait_async0();
  __syncthreads();

  // S = q @ k^T : wave handles 16 rows x 64 cols
  const int m0 = wave * 16;
  {
    const v16bf aq = load_a_frag(&qs[m0 + (lane & 15)][0], lane);
#pragma unroll
    for (int nt = 0; nt < 4; ++nt) {
      const v16bf bk = load_b_frag(&ks[nt * 16 + (lane & 15)][0], lane);
      v8f a1;
#pragma unroll
      for (int r = 0; r < 8; ++r) a1[r] = 0.0f;
      a1 = __builtin_amdgcn_wmma_f32_16x16x32_bf16(false, aq, false, bk,
                                                   (short)0, a1, false, false);
#pragma unroll
      for (int r = 0; r < 8; ++r)
        att[m0 + ((lane < 16) ? r : r + 8)][nt * 16 + (lane & 15)] = a1[r];
    }
  }
  __syncthreads();

  // scale + bias + mask + softmax (row-parallel)
  if (tid < 64) {
    const float scale = 0.17677669529663688f;            // 32^-0.5
    const int pi = tid >> 3, pj = tid & 7;
    const int cp = cnt_s[tid];
    float mx = -3.0e38f;
    for (int q = 0; q < 64; ++q) {
      const int qi = q >> 3, qj = q & 7;
      float a = att[tid][q] * scale +
                rpb[((pi - qi + 7) * 15 + (pj - qj + 7)) * 8 + head];
      if (shift > 0 && cp != cnt_s[q]) a -= 100.0f;
      att[tid][q] = a;
      mx = fmaxf(mx, a);
    }
    float sum = 0.0f;
    for (int q = 0; q < 64; ++q) {
      const float e = __expf(att[tid][q] - mx);
      att[tid][q] = e;
      sum += e;
    }
    const float inv = 1.0f / sum;
    for (int q = 0; q < 64; ++q) attb[tid][q] = bfc(att[tid][q] * inv);
  }
  __syncthreads();

  // O = P @ v : M=64 x N=32 x K=64 (two K-steps)
  v8f acc2[2];
#pragma unroll
  for (int nt = 0; nt < 2; ++nt)
#pragma unroll
    for (int r = 0; r < 8; ++r) acc2[nt][r] = 0.0f;
#pragma unroll
  for (int kk = 0; kk < 64; kk += 32) {
    const v16bf aa = load_a_frag(&attb[m0 + (lane & 15)][kk], lane);
#pragma unroll
    for (int nt = 0; nt < 2; ++nt) {
      const v16bf bv = load_b_frag(&vts[nt * 16 + (lane & 15)][kk], lane);
      acc2[nt] = __builtin_amdgcn_wmma_f32_16x16x32_bf16(
          false, aa, false, bv, (short)0, acc2[nt], false, false);
    }
  }
#pragma unroll
  for (int nt = 0; nt < 2; ++nt)
#pragma unroll
    for (int r = 0; r < 8; ++r) {
      const int pr = m0 + ((lane < 16) ? r : r + 8);
      const int d  = nt * 16 + (lane & 15);
      outp[(size_t)tok_s[pr] * 256 + head * 32 + d] = bfc(acc2[nt][r]);
    }
}

// ---------------------------------------------------------------------------
// Host orchestration
// ---------------------------------------------------------------------------
extern "C" void kernel_launch(void* const* d_in, const int* in_sizes, int n_in,
                              void* d_out, int out_size, void* d_ws,
                              size_t ws_size, hipStream_t stream) {
  const float* x     = (const float*)d_in[0];
  const float* cv1w  = (const float*)d_in[1];
  const float* cv1bn = (const float*)d_in[2];
  const float* cv2w  = (const float*)d_in[3];
  const float* cv2bn = (const float*)d_in[4];
  const float* cv3w  = (const float*)d_in[5];
  const float* cv3bn = (const float*)d_in[6];
  const float* n1g   = (const float*)d_in[7];
  const float* n1b   = (const float*)d_in[8];
  const float* qkvw  = (const float*)d_in[9];
  const float* qkvb  = (const float*)d_in[10];
  const float* projw = (const float*)d_in[11];
  const float* projb = (const float*)d_in[12];
  const float* rpb   = (const float*)d_in[13];
  const float* n2g   = (const float*)d_in[14];
  const float* n2b   = (const float*)d_in[15];
  const float* fc1w  = (const float*)d_in[16];
  const float* fc1b  = (const float*)d_in[17];
  const float* fc2w  = (const float*)d_in[18];
  const float* fc2b  = (const float*)d_in[19];
  float* out = (float*)d_out;

  if (ws_size < 440401920ull) return;   // ~420 MB scratch
  float*  X    = (float*)d_ws;                       // tokens fp32 (65536,256)
  float*  X1   = X + (size_t)16777216;               // ping-pong fp32
  __bf16* TMPb = (__bf16*)(X1 + (size_t)16777216);   // LN out / attn out bf16
  __bf16* BIGb = TMPb + (size_t)16777216;            // qkv(768)/fc1(1024) bf16
  __bf16* Y2b  = BIGb + (size_t)67108864;            // cv2 planar bf16
  __bf16* XPb  = Y2b + (size_t)16777216;             // x -> bf16 planar
  __bf16* XTb  = XPb + (size_t)33554432;             // fc2 out bf16 tokens
  __bf16* WA   = XTb + (size_t)16777216;             // weight arena (4MB)
  __bf16* Wc1  = WA;
  __bf16* Wc2  = Wc1 + 131072;
  __bf16* Wc3  = Wc2 + 131072;
  __bf16* Wqkv = Wc3 + 262144;
  __bf16* Wprj = Wqkv + 393216;
  __bf16* Wf1  = Wprj + 131072;
  __bf16* Wf2  = Wf1 + 524288;

  auto f2b = [&](const float* src, __bf16* dst, long n) {
    hipLaunchKernelGGL(f2b_kernel, dim3(1024), dim3(256), 0, stream, src, dst, n);
  };
  f2b(x, XPb, 33554432L);
  f2b(cv1w, Wc1, 131072L);
  f2b(cv2w, Wc2, 131072L);
  f2b(cv3w, Wc3, 262144L);
  f2b(qkvw, Wqkv, 393216L);
  f2b(projw, Wprj, 131072L);
  f2b(fc1w, Wf1, 524288L);
  f2b(fc2w, Wf2, 524288L);

  const dim3 blk(256);
  auto gemm = [&](const __bf16* A, const __bf16* B0, const __bf16* B1,
                  float* Cf, __bf16* Cb, const float* res, const float* e0,
                  int mode, int bpath, int M, int N, int K, int Ksplit,
                  int Mtot, int batches, long b0b, long b0k, long b0n,
                  long b1b, long b1k, long b1n, long cb, long cm, long cn) {
    GemmP p{A, B0, B1, Cf, Cb, res, e0, b0b, b0k, b0n, b1b, b1k, b1n,
            cb, cm, cn, M, N, K, Ksplit, Mtot, mode, bpath};
    dim3 grid(N / 128, M / 128, batches);
    hipLaunchKernelGGL(gemm_wmma_bf16, grid, blk, 0, stream, p);
  };

  // cv1: XPb planar -> X fp32 token-major, BN+SiLU
  gemm(Wc1, XPb, nullptr, X, nullptr, nullptr, cv1bn, 0, 1,
       256, 4096, 512, 0, 256, 16,
       (long)512 * 4096, 4096, 1, 0, 0, 0, (long)4096 * 256, 1, 256);
  // cv2: XPb planar -> Y2b bf16 planar, BN+SiLU
  gemm(Wc2, XPb, nullptr, nullptr, Y2b, nullptr, cv2bn, 0, 1,
       256, 4096, 512, 0, 256, 16,
       (long)512 * 4096, 4096, 1, 0, 0, 0, (long)256 * 4096, 4096, 1);

  for (int i = 0; i < 2; ++i) {
    const int shift = (i == 0) ? 0 : 4;
    hipLaunchKernelGGL(ln_kernel, dim3(65536), dim3(256), 0, stream, X, TMPb,
                       n1g + i * 256, n1b + i * 256);
    // qkv = LN(X) @ W^T + b -> BIGb bf16 (65536,768)
    gemm(Wqkv + (size_t)i * 768 * 256, TMPb, nullptr, nullptr, BIGb, nullptr,
         qkvb + i * 768, 1, 0, 768, 65536, 256, 0, 768, 1,
         0, 1, 256, 0, 0, 0, 0, 1, 768);
    // attention -> TMPb bf16 (65536,256)
    hipLaunchKernelGGL(swin_attn, dim3(1024, 8), dim3(128), 0, stream, BIGb,
                       TMPb, rpb + (size_t)i * 1800, shift);
    // X1 = X + attn @ proj^T + b   (fp32)
    gemm(Wprj + (size_t)i * 65536, TMPb, nullptr, X1, nullptr, X,
         projb + i * 256, 3, 0, 256, 65536, 256, 0, 256, 1,
         0, 1, 256, 0, 0, 0, 0, 1, 256);
    hipLaunchKernelGGL(ln_kernel, dim3(65536), dim3(256), 0, stream, X1, TMPb,
                       n2g + i * 256, n2b + i * 256);
    // H = GELU(LN(X1) @ fc1^T + b1) -> BIGb bf16 (65536,1024)
    gemm(Wf1 + (size_t)i * 262144, TMPb, nullptr, nullptr, BIGb, nullptr,
         fc1b + i * 1024, 2, 0, 1024, 65536, 256, 0, 1024, 1,
         0, 1, 256, 0, 0, 0, 0, 1, 1024);
    // X = X1 + H @ fc2^T + b2  (fp32 + bf16 copy for cv3)
    gemm(Wf2 + (size_t)i * 262144, BIGb, nullptr, X, XTb, X1,
         fc2b + i * 256, 3, 0, 256, 65536, 1024, 0, 256, 1,
         0, 1, 1024, 0, 0, 0, 0, 1, 256);
  }

  // cv3 over concat([XTb tokens ch0..255, Y2b planar ch256..511]) -> out
  gemm(Wc3, XTb, Y2b, out, nullptr, nullptr, cv3bn, 0, 2,
       512, 4096, 512, 256, 512, 16,
       (long)4096 * 256, 1, 256,
       (long)256 * 4096, 4096, 1,
       (long)512 * 4096, 4096, 1);
}
